// RNN_46548855554081
// MI455X (gfx1250) — matrix operand order repaired and checked by reading, not compiled, not run on hardware
//
#include <hip/hip_runtime.h>
#include <math.h>

typedef __attribute__((ext_vector_type(16))) _Float16 v16h;
typedef __attribute__((ext_vector_type(8)))  _Float16 v8h;
typedef __attribute__((ext_vector_type(8)))  float    v8f;

#define IN_  64
#define H_   1024
#define B_   64
#define T_   512
#define SCAN_BLOCKS 8
#define SCAN_LDS_BYTES (128 * H_ * 2)   // 128 hidden rows x 1024 k x f16 = 256 KB

// ---------------------------------------------------------------------------
// WMMA fragment helpers for v_wmma_f32_16x16x32_f16 (wave32); A/B below use this code's contiguous k order, not the ISA table order
//  A (16x32 f16): lane l holds row l%16; 16 contiguous halves at
//                 k = (l<16 ? 0 : 16).
//  B (32x16 f16): lane l holds column l%16; 16 contiguous halves at
//                 k = (l<16 ? 0 : 16)  -> direct loads from N x K row-major W.
//  C/D (16x16 f32): vgpr r -> row r + (l<16?0:8), col l%16.
// ---------------------------------------------------------------------------
__device__ __forceinline__ v16h combine_a(v8h lo, v8h hi) {
    v16h r;
#pragma unroll
    for (int i = 0; i < 8; ++i) { r[i] = lo[i]; r[i + 8] = hi[i]; }
    return r;
}

__device__ __forceinline__ v8f wmma_f16(v16h a, v16h b, v8f c) {
    return __builtin_amdgcn_wmma_f32_16x16x32_f16(false, a, false, b,
                                                  (short)0, c, false, false);
}

// ---------------------------------------------------------------------------
// Utility kernels
// ---------------------------------------------------------------------------
__global__ void f32_to_f16_kernel(const float* __restrict__ s,
                                  _Float16* __restrict__ d, int n) {
    int i = blockIdx.x * blockDim.x + threadIdx.x;
    if (i < n) d[i] = (_Float16)s[i];
}

__global__ void init_cnt_kernel(unsigned* c) {
    if (threadIdx.x < 2) c[threadIdx.x] = 0u;
}

// ---------------------------------------------------------------------------
// GEMM with bias: C(MxN, f32) = A(MxK, f16, lda) @ Bt(NxK, f16, ldb)^T + biases
// One wave = 16(M) x 64(N) strip, software-pipelined K loop (double-buffered
// fragments so VMEM overlaps the matrix pipe instead of s_wait_loadcnt 0).
// ---------------------------------------------------------------------------
__global__ void gemm_bias_kernel(const _Float16* __restrict__ A, int lda,
                                 const _Float16* __restrict__ Bt, int ldb,
                                 const float* __restrict__ bias0,
                                 const float* __restrict__ bias1,
                                 float* __restrict__ C, int ldc,
                                 int M, int N, int K) {
    int lane = threadIdx.x & 31;
    size_t wt = ((size_t)blockIdx.x * blockDim.x + threadIdx.x) >> 5;
    int mtiles  = M >> 4;
    int ngroups = N >> 6;
    if (wt >= (size_t)mtiles * ngroups) return;
    int m0 = (int)(wt % mtiles) * 16;
    int n0 = (int)(wt / mtiles) * 64;

    // per-lane base pointers (bumped by +32 halves per k-chunk)
    const _Float16* pa  = A  + (size_t)(m0 + (lane & 15)) * lda + ((lane & 16) ? 16 : 0);
    const _Float16* pb0 = Bt + (size_t)(n0 +      (lane & 15)) * ldb + ((lane & 16) ? 16 : 0);
    const _Float16* pb1 = pb0 + (size_t)16 * ldb;
    const _Float16* pb2 = pb0 + (size_t)32 * ldb;
    const _Float16* pb3 = pb0 + (size_t)48 * ldb;

    v8f acc[4];
#pragma unroll
    for (int j = 0; j < 4; ++j) acc[j] = (v8f)0.0f;

    v8h  alo = *(const v8h*)(pa), ahi = *(const v8h*)(pa + 8);
    v16h b0 = *(const v16h*)(pb0), b1 = *(const v16h*)(pb1);
    v16h b2 = *(const v16h*)(pb2), b3 = *(const v16h*)(pb3);

#pragma unroll 2
    for (int kc = 0; kc < K - 32; kc += 32) {
        // prefetch next chunk into fresh registers before consuming current
        v8h  nlo = *(const v8h*)(pa + 32), nhi = *(const v8h*)(pa + 40);
        v16h nb0 = *(const v16h*)(pb0 + 32), nb1 = *(const v16h*)(pb1 + 32);
        v16h nb2 = *(const v16h*)(pb2 + 32), nb3 = *(const v16h*)(pb3 + 32);
        v16h a = combine_a(alo, ahi);
        acc[0] = wmma_f16(a, b0, acc[0]);
        acc[1] = wmma_f16(a, b1, acc[1]);
        acc[2] = wmma_f16(a, b2, acc[2]);
        acc[3] = wmma_f16(a, b3, acc[3]);
        alo = nlo; ahi = nhi; b0 = nb0; b1 = nb1; b2 = nb2; b3 = nb3;
        pa += 32; pb0 += 32; pb1 += 32; pb2 += 32; pb3 += 32;
    }
    {   // final chunk
        v16h a = combine_a(alo, ahi);
        acc[0] = wmma_f16(a, b0, acc[0]);
        acc[1] = wmma_f16(a, b1, acc[1]);
        acc[2] = wmma_f16(a, b2, acc[2]);
        acc[3] = wmma_f16(a, b3, acc[3]);
    }

    int col_lo = lane & 15;
    int row_hi = (lane & 16) ? 8 : 0;
#pragma unroll
    for (int j = 0; j < 4; ++j) {
        int col = n0 + j * 16 + col_lo;
        float bv = bias0[col] + bias1[col];
#pragma unroll
        for (int r = 0; r < 8; ++r) {
            int row = m0 + r + row_hi;
            C[(size_t)row * ldc + col] = acc[j][r] + bv;
        }
    }
}

// ---------------------------------------------------------------------------
// Persistent RNN scan: 8 blocks (one per WGP) x 256 threads = 64 waves; each
// wave owns a 16(batch) x 64(hidden) tile of the state.  Each block stages its
// 128x1024 f16 W_hh slice (256 KB) into LDS ONCE; all 512 steps then read
// B-fragments via ds_load while only the hidden state streams from L2.
//   h_new = tanh(xp[:,t,:] + h @ Whh^T)   + grid-wide barrier per step.
// ---------------------------------------------------------------------------
__device__ __forceinline__ void grid_sync(unsigned* cnt, unsigned target) {
    __threadfence();
    __syncthreads();
    if (threadIdx.x == 0) {
        atomicAdd(cnt, 1u);
        while (atomicAdd(cnt, 0u) < target)
            __builtin_amdgcn_s_sleep(2);
    }
    __syncthreads();
    __threadfence();
}

__global__ void rnn_scan_kernel(const float* __restrict__ xp,        // (B*T, H)
                                const _Float16* __restrict__ Whh,    // (H, H)
                                const _Float16* __restrict__ h_init, // (B, H) f16
                                _Float16* __restrict__ r_out,        // (B*T,H) or 0
                                _Float16* __restrict__ hping,        // (2,B,H) or 0
                                float* __restrict__ r2t0,            // (B,H) or 0
                                float* __restrict__ h_final,         // (B,H)
                                unsigned* __restrict__ cnt) {
    extern __shared__ _Float16 lds_w[];          // 128 x 1024 f16 (256 KB)

    int lane = threadIdx.x & 31;
    int wv   = threadIdx.x >> 5;                 // 0..7
    int m0   = (wv & 3) * 16;                    // batch tile
    int n0l  = (wv >> 2) * 64;                   // hidden tile, block-local
    int nbase = blockIdx.x * 128;                // block's hidden slice

    // ---- stage W slice into LDS (global_load_b128 -> ds_store_b128) ----
    {
        const float4* src = (const float4*)(Whh + (size_t)nbase * H_);
        float4* dst = (float4*)lds_w;
        for (int i = threadIdx.x; i < SCAN_LDS_BYTES / 16; i += blockDim.x)
            dst[i] = src[i];
    }
    __syncthreads();

    int col_lo = lane & 15;
    int row_hi = (lane & 16) ? 8 : 0;
    // block-local LDS offsets for the 4 B-fragment columns (advance +32/chunk)
    unsigned boff = (unsigned)(n0l + col_lo) * H_ + ((lane & 16) ? 16 : 0);

#pragma unroll 1
    for (int t = 0; t < T_; ++t) {
        const _Float16* Asrc;
        size_t lda;
        if (t == 0)     { Asrc = h_init;                                  lda = H_; }
        else if (r_out) { Asrc = r_out + (size_t)(t - 1) * H_;            lda = (size_t)T_ * H_; }
        else            { Asrc = hping + (size_t)((t - 1) & 1) * B_ * H_; lda = H_; }

        const _Float16* pa = Asrc + (size_t)(m0 + col_lo) * lda + ((lane & 16) ? 16 : 0);

        v8f acc[4];
#pragma unroll
        for (int j = 0; j < 4; ++j) acc[j] = (v8f)0.0f;

        v8h alo = *(const v8h*)(pa), ahi = *(const v8h*)(pa + 8);
#pragma unroll 2
        for (int kc = 0; kc < H_ - 32; kc += 32) {
            v8h nlo = *(const v8h*)(pa + 32), nhi = *(const v8h*)(pa + 40);
            v16h a = combine_a(alo, ahi);
            acc[0] = wmma_f16(a, *(const v16h*)&lds_w[boff              + kc], acc[0]);
            acc[1] = wmma_f16(a, *(const v16h*)&lds_w[boff + 16u * H_   + kc], acc[1]);
            acc[2] = wmma_f16(a, *(const v16h*)&lds_w[boff + 32u * H_   + kc], acc[2]);
            acc[3] = wmma_f16(a, *(const v16h*)&lds_w[boff + 48u * H_   + kc], acc[3]);
            alo = nlo; ahi = nhi; pa += 32;
        }
        {
            const int kc = H_ - 32;
            v16h a = combine_a(alo, ahi);
            acc[0] = wmma_f16(a, *(const v16h*)&lds_w[boff              + kc], acc[0]);
            acc[1] = wmma_f16(a, *(const v16h*)&lds_w[boff + 16u * H_   + kc], acc[1]);
            acc[2] = wmma_f16(a, *(const v16h*)&lds_w[boff + 32u * H_   + kc], acc[2]);
            acc[3] = wmma_f16(a, *(const v16h*)&lds_w[boff + 48u * H_   + kc], acc[3]);
        }

#pragma unroll
        for (int j = 0; j < 4; ++j) {
            int col = nbase + n0l + j * 16 + col_lo;
#pragma unroll
            for (int r = 0; r < 8; ++r) {
                int row = m0 + r + row_hi;          // batch index
                float v = tanhf(acc[j][r] + xp[((size_t)row * T_ + t) * H_ + col]);
                if (r_out) {
                    r_out[((size_t)row * T_ + t) * H_ + col] = (_Float16)v;
                } else {
                    hping[(size_t)(t & 1) * B_ * H_ + (size_t)row * H_ + col] = (_Float16)v;
                    if (t == 0) r2t0[(size_t)row * H_ + col] = v;
                }
                if (t == T_ - 1) h_final[(size_t)row * H_ + col] = v;
            }
        }
        grid_sync(cnt, (unsigned)(t + 1) * SCAN_BLOCKS);
    }
}

// out[b] = dot(r2[:,0,:][b], W_out) + b_out   (tiny: 64 x 1024)
__global__ void proj_kernel(const float* __restrict__ r2t0,
                            const float* __restrict__ Wout,
                            const float* __restrict__ bout,
                            float* __restrict__ out) {
    int b = threadIdx.x;
    if (b >= B_) return;
    float s = bout[0];
    for (int h = 0; h < H_; ++h) s += r2t0[(size_t)b * H_ + h] * Wout[h];
    out[b] = s;
}

// ---------------------------------------------------------------------------
extern "C" void kernel_launch(void* const* d_in, const int* in_sizes, int n_in,
                              void* d_out, int out_size, void* d_ws, size_t ws_size,
                              hipStream_t stream) {
    (void)in_sizes; (void)n_in; (void)out_size; (void)ws_size;
    const float* x       = (const float*)d_in[0];   // (B,T,IN)
    const float* h_state = (const float*)d_in[1];   // (2,B,H)
    const float* W_ih0   = (const float*)d_in[2];   // (H,IN)
    const float* W_hh0   = (const float*)d_in[3];   // (H,H)
    const float* b_ih0   = (const float*)d_in[4];
    const float* b_hh0   = (const float*)d_in[5];
    const float* W_ih1   = (const float*)d_in[6];   // (H,H)
    const float* W_hh1   = (const float*)d_in[7];   // (H,H)
    const float* b_ih1   = (const float*)d_in[8];
    const float* b_hh1   = (const float*)d_in[9];
    const float* W_out   = (const float*)d_in[10];  // (1,H)
    const float* b_out   = (const float*)d_in[11];

    float* out    = (float*)d_out;                  // [0,64): proj
    float* h1_out = (float*)d_out + 64;             // (B,H)
    float* h2_out = (float*)d_out + 64 + B_ * H_;   // (B,H)

    // workspace bump allocator (256B aligned)
    char* w = (char*)d_ws;
    size_t off = 0;
    auto alloc = [&](size_t bytes) -> void* {
        void* p = w + off;
        off = (off + bytes + 255) & ~(size_t)255;
        return p;
    };
    const size_t MT = (size_t)B_ * T_;              // 32768 rows
    _Float16* x16    = (_Float16*)alloc(MT * IN_ * 2);
    _Float16* wih0_h = (_Float16*)alloc((size_t)H_ * IN_ * 2);
    _Float16* whh0_h = (_Float16*)alloc((size_t)H_ * H_ * 2);
    _Float16* wih1_h = (_Float16*)alloc((size_t)H_ * H_ * 2);
    _Float16* whh1_h = (_Float16*)alloc((size_t)H_ * H_ * 2);
    _Float16* h1init = (_Float16*)alloc((size_t)B_ * H_ * 2);
    _Float16* h2init = (_Float16*)alloc((size_t)B_ * H_ * 2);
    float*    xp     = (float*)alloc(MT * H_ * 4);  // reused for xp0 then xp1
    _Float16* r1     = (_Float16*)alloc(MT * H_ * 2);
    _Float16* h2ping = (_Float16*)alloc((size_t)2 * B_ * H_ * 2);
    float*    r2t0   = (float*)alloc((size_t)B_ * H_ * 4);
    unsigned* cnt    = (unsigned*)alloc(256);

    init_cnt_kernel<<<1, 32, 0, stream>>>(cnt);

    auto cvt = [&](const float* s, _Float16* d, int n) {
        f32_to_f16_kernel<<<(n + 255) / 256, 256, 0, stream>>>(s, d, n);
    };
    cvt(x,       x16,    (int)(MT * IN_));
    cvt(W_ih0,   wih0_h, H_ * IN_);
    cvt(W_hh0,   whh0_h, H_ * H_);
    cvt(W_ih1,   wih1_h, H_ * H_);
    cvt(W_hh1,   whh1_h, H_ * H_);
    cvt(h_state,           h1init, B_ * H_);
    cvt(h_state + B_ * H_, h2init, B_ * H_);

    // xp0 = x @ W_ih0^T + b_ih0 + b_hh0     (M=32768, N=1024, K=64)
    {
        size_t waves = (MT / 16) * (H_ / 64);
        int blocks = (int)((waves * 32 + 255) / 256);
        gemm_bias_kernel<<<blocks, 256, 0, stream>>>(x16, IN_, wih0_h, IN_,
                                                     b_ih0, b_hh0, xp, H_,
                                                     (int)MT, H_, IN_);
    }
    // layer-1 scan -> r1 (f16), h1_out (f32)
    rnn_scan_kernel<<<SCAN_BLOCKS, 256, SCAN_LDS_BYTES, stream>>>(
        xp, whh0_h, h1init, r1, nullptr, nullptr, h1_out, cnt + 0);
    // xp1 = r1 @ W_ih1^T + b_ih1 + b_hh1    (M=32768, N=1024, K=1024)
    {
        size_t waves = (MT / 16) * (H_ / 64);
        int blocks = (int)((waves * 32 + 255) / 256);
        gemm_bias_kernel<<<blocks, 256, 0, stream>>>(r1, H_, wih1_h, H_,
                                                     b_ih1, b_hh1, xp, H_,
                                                     (int)MT, H_, H_);
    }
    // layer-2 scan -> h2_out (f32), r2[:,0,:] capture
    rnn_scan_kernel<<<SCAN_BLOCKS, 256, SCAN_LDS_BYTES, stream>>>(
        xp, whh1_h, h2init, nullptr, h2ping, r2t0, h2_out, cnt + 1);
    // out = r2[:,0,:] @ W_out^T + b_out
    proj_kernel<<<1, 64, 0, stream>>>(r2t0, W_out, b_out, out);
}
